// MetapathEmbed_78683800863346
// MI455X (gfx1250) — compile-verified, matrix operand order checked by reading, not run on hardware
//
#include <hip/hip_runtime.h>
#include <hip/hip_bf16.h>

// MetapathEmbed on MI455X (gfx1250, wave32, WMMA bf16 16x16x32).
//
// out = ((batch_pools @ metapath^T) @ (card_embeddings @ kernel))
// Dominant GEMM: P = batch_pools(128x16384) @ metapath^T(16384x16384)
//   -> memory-bound on the 1 GiB metapath read (~46 us at 23.3 TB/s).
// fp32 precision via bf16x3 split (hi*hi + hi*lo + lo*hi).
// metapath tiles staged to LDS with GLOBAL_LOAD_ASYNC_TO_LDS_B128 (ASYNCcnt),
// split f32->bf16 hi/lo once per workgroup, consumed via ds_load_b128.
// N-tile = 32 (two 16x16 WMMA tiles per wave) to halve A-plane L2 traffic.

typedef __attribute__((ext_vector_type(16))) __bf16 v16bf;
typedef __attribute__((ext_vector_type(8)))  __bf16 v8bf;
typedef __attribute__((ext_vector_type(2)))  __bf16 v2bf;
typedef __attribute__((ext_vector_type(8)))  float  v8f;
typedef __attribute__((ext_vector_type(4)))  float  v4f;
typedef __attribute__((ext_vector_type(2)))  float  v2f;

#define WMMA_BF16(a, b, c)                                                     \
  __builtin_amdgcn_wmma_f32_16x16x32_bf16(false, (a), false, (b), (short)0,    \
                                          (c), false, false)

// Low 32 bits of a generic pointer to LDS == LDS byte address (ISA 10.2).
__device__ __forceinline__ uint32_t lds_lo32(const void* p) {
  return (uint32_t)(uintptr_t)p;
}

// GV-mode async DMA: LDS[lds_addr..+15] = MEM[gaddr..+15]; tracked by ASYNCcnt.
__device__ __forceinline__ void async_g2l_b128(uint32_t lds_addr,
                                               const void* g) {
  unsigned long long ga = (unsigned long long)(uintptr_t)g;
  asm volatile("global_load_async_to_lds_b128 %0, %1, off" ::"v"(lds_addr),
               "v"(ga)
               : "memory");
}
__device__ __forceinline__ void wait_asynccnt0() {
  asm volatile("s_wait_asynccnt 0x0" ::: "memory");
}

__device__ __forceinline__ v16bf cat16(v8bf a, v8bf b) {
  return __builtin_shufflevector(a, b, 0, 1, 2, 3, 4, 5, 6, 7, 8, 9, 10, 11,
                                 12, 13, 14, 15);
}

// ---------------------------------------------------------------------------
// Pass 0: split fp32 -> (bf16 hi, bf16 lo) planes.  x ~= hi + lo.
// ---------------------------------------------------------------------------
__global__ __launch_bounds__(256) void split_f32_to_bf16x2(
    const float* __restrict__ in, __bf16* __restrict__ hi,
    __bf16* __restrict__ lo, int n) {
  int i = blockIdx.x * blockDim.x + threadIdx.x;
  if (i < n) {
    float x = in[i];
    __bf16 h = (__bf16)x;
    hi[i] = h;
    lo[i] = (__bf16)(x - (float)h);
  }
}

// ---------------------------------------------------------------------------
// Pass 1: Xt[m][c] = sum_e emb[c][e] * kern[e][m]  (transposed X, fp32)
// ---------------------------------------------------------------------------
__global__ __launch_bounds__(256) void xt_gemm_small(
    const float* __restrict__ emb,   // [16384][64]
    const float* __restrict__ kern,  // [64][128]
    float* __restrict__ Xt) {        // [128][16384]
  __shared__ float kl[64 * 128];
  for (int i = threadIdx.x; i < 64 * 128; i += 256) kl[i] = kern[i];
  __syncthreads();

  const int c = blockIdx.x * 256 + threadIdx.x;
  float er[64];
#pragma unroll
  for (int e = 0; e < 64; ++e) er[e] = emb[(size_t)c * 64 + e];

  for (int m = 0; m < 128; ++m) {
    float s = 0.f;
#pragma unroll
    for (int e = 0; e < 64; ++e) s = fmaf(er[e], kl[e * 128 + m], s);
    Xt[(size_t)m * 16384 + c] = s;
  }
}

// ---------------------------------------------------------------------------
// Pass 2 (dominant): C[128][N] = A[128][K] * Bt[N][K]^T, C written split.
// Block = 256 thr = 8 waves; wave w -> rows 16w..16w+15; block -> 32 cols
// (two 16x16 WMMA tiles / wave -> A fragments amortized over 6 WMMAs).
// Per K-chunk (KC=128): async-DMA the 32xKC f32 Bt tile into LDS (double
// buffered), one shared f32->bf16 hi/lo conversion, waves consume fragments
// via ds_load_b128.  A is pre-split bf16 (L2-resident), loaded from global.
// ---------------------------------------------------------------------------
constexpr int KC = 128;     // K-chunk
constexpr int RS = KC + 8;  // padded bf16 row stride: 272 B, 16B-aligned,
                            // 68 dwords -> 4-bank rotation/row (no conflicts)
constexpr int NT = 32;      // column tile per workgroup

__global__ __launch_bounds__(256) void gemm_metapath_async(
    const __bf16* __restrict__ Ahi, const __bf16* __restrict__ Alo,
    const float* __restrict__ Bt, __bf16* __restrict__ Chi,
    __bf16* __restrict__ Clo, int N, int K) {
  __shared__ float  rawb[2][NT * KC];   // 32 KB raw f32 tiles
  __shared__ __bf16 bhiL[2][NT * RS];   // 17 KB hi planes
  __shared__ __bf16 bloL[2][NT * RS];   // 17 KB lo planes

  const int tid  = threadIdx.x;
  const int lane = tid & 31;
  const int wave = tid >> 5;   // 0..7: output row block
  const int hlf  = lane >> 4;  // 0: K 0..15 of step, 1: K 16..31
  const int l16  = lane & 15;
  const int n0   = blockIdx.x * NT;
  const int NCH  = K / KC;

  // Issue the async DMA for one chunk: NT rows x KC f32 = 16 KB, 4 rounds of
  // one b128 per thread (256 thr x 16 B = 4 KB/round).
  auto stage = [&](int chunk, int buf) {
    const int d0 = chunk * KC;
#pragma unroll
    for (int r = 0; r < 4; ++r) {
      const int f   = (r * 256 + tid) * 4;  // dword index within NTxKC tile
      const int row = f >> 7;               // f / KC
      const int col = f & (KC - 1);
      async_g2l_b128(lds_lo32(&rawb[buf][f]),
                     Bt + (size_t)(n0 + row) * K + d0 + col);
    }
  };

  // Split one raw chunk into bf16 hi/lo planes; each element touched once.
  auto convert = [&](int buf) {
#pragma unroll
    for (int j = 0; j < 8; ++j) {
      const int p   = tid + (j << 8);  // pair index 0..2047
      const int row = p >> 6;          // 64 pairs per row
      const int col = (p & 63) * 2;
      v2f x = *(const v2f*)&rawb[buf][p * 2];
      __bf16 h0 = (__bf16)x[0], h1 = (__bf16)x[1];
      v2bf hh = {h0, h1};
      v2bf ll = {(__bf16)(x[0] - (float)h0), (__bf16)(x[1] - (float)h1)};
      *(v2bf*)&bhiL[buf][row * RS + col] = hh;
      *(v2bf*)&bloL[buf][row * RS + col] = ll;
    }
  };

  // A fragment (16x32 bf16): lane l<16 -> row M=l, K 0..15 ; l>=16 -> K 16..31
  const size_t arow = (size_t)(wave * 16 + l16);
  const __bf16* aph = Ahi + arow * K + hlf * 16;
  const __bf16* apl = Alo + arow * K + hlf * 16;

  v8f acc0 = {0.f, 0.f, 0.f, 0.f, 0.f, 0.f, 0.f, 0.f};
  v8f acc1 = {0.f, 0.f, 0.f, 0.f, 0.f, 0.f, 0.f, 0.f};

  // Software pipeline: DMA chunk c+1 overlaps compute of chunk c.
  stage(0, 0);
  wait_asynccnt0();
  __syncthreads();
  convert(0);
  if (NCH > 1) stage(1, 1);
  __syncthreads();

  for (int c = 0; c < NCH; ++c) {
    const int buf = c & 1;
    // B fragments (32x16): lane l<16 -> col N=l, K 0..15 ; l>=16 -> K 16..31.
    // Column block 0 -> LDS rows l16, block 1 -> LDS rows 16+l16.
    const __bf16* bh0 = &bhiL[buf][l16 * RS + hlf * 16];
    const __bf16* bl0 = &bloL[buf][l16 * RS + hlf * 16];
    const __bf16* bh1 = bh0 + 16 * RS;
    const __bf16* bl1 = bl0 + 16 * RS;
    const int dg = c * KC;
#pragma unroll
    for (int ks = 0; ks < KC / 32; ++ks) {
      const int d = ks * 32;
      v16bf ah = *(const v16bf*)(aph + dg + d);
      v16bf al = *(const v16bf*)(apl + dg + d);
      v16bf bh0v = cat16(*(const v8bf*)(bh0 + d), *(const v8bf*)(bh0 + d + 8));
      v16bf bl0v = cat16(*(const v8bf*)(bl0 + d), *(const v8bf*)(bl0 + d + 8));
      v16bf bh1v = cat16(*(const v8bf*)(bh1 + d), *(const v8bf*)(bh1 + d + 8));
      v16bf bl1v = cat16(*(const v8bf*)(bl1 + d), *(const v8bf*)(bl1 + d + 8));
      // Two independent accumulator chains (ILP across column blocks).
      acc0 = WMMA_BF16(ah, bh0v, acc0);  // hi * hi
      acc1 = WMMA_BF16(ah, bh1v, acc1);
      acc0 = WMMA_BF16(ah, bl0v, acc0);  // hi * lo
      acc1 = WMMA_BF16(ah, bl1v, acc1);
      acc0 = WMMA_BF16(al, bh0v, acc0);  // lo * hi
      acc1 = WMMA_BF16(al, bh1v, acc1);
    }
    if (c + 1 < NCH) {
      wait_asynccnt0();  // raw[(c+1)&1] DMA complete (per wave)
      __syncthreads();   // ... for every wave
      convert((c + 1) & 1);
      if (c + 2 < NCH) stage(c + 2, c & 1);  // raw[c&1] consumed last iter
      __syncthreads();   // conv[(c+1)&1] published
    }
  }

  // C/D 16x16 f32 layout: VGPR v -> M = v + 8*hlf, N = l16.  Write split.
#pragma unroll
  for (int v = 0; v < 8; ++v) {
    const int m = wave * 16 + v + hlf * 8;
    const size_t idx = (size_t)m * N + n0 + l16;
    float v0 = acc0[v], v1 = acc1[v];
    __bf16 h0 = (__bf16)v0, h1 = (__bf16)v1;
    Chi[idx] = h0;
    Clo[idx] = (__bf16)(v0 - (float)h0);
    Chi[idx + 16] = h1;
    Clo[idx + 16] = (__bf16)(v1 - (float)h1);
  }
}

// ---------------------------------------------------------------------------
// Pass 3 (small): Cout[128][N] = A[128][K] * Bt[N][K]^T, fp32 out, direct
// global loads (B split on the fly in VALU).
// ---------------------------------------------------------------------------
__global__ __launch_bounds__(256) void gemm_out_bf16x3(
    const __bf16* __restrict__ Ahi, const __bf16* __restrict__ Alo,
    const float* __restrict__ Bt, float* __restrict__ Cout, int N, int K) {
  const int lane = threadIdx.x & 31;
  const int wave = threadIdx.x >> 5;
  const int hlf  = lane >> 4;
  const int l16  = lane & 15;
  const int n0   = blockIdx.x * 16;

  const size_t arow = (size_t)(wave * 16 + l16);
  const __bf16* aph = Ahi + arow * K + hlf * 16;
  const __bf16* apl = Alo + arow * K + hlf * 16;
  const float*  bp  = Bt + (size_t)(n0 + l16) * K + hlf * 16;

  v8f acc = {0.f, 0.f, 0.f, 0.f, 0.f, 0.f, 0.f, 0.f};

  for (int d0 = 0; d0 < K; d0 += 32) {
    v16bf ah = *(const v16bf*)(aph + d0);
    v16bf al = *(const v16bf*)(apl + d0);
    v4f q0 = ((const v4f*)(bp + d0))[0];
    v4f q1 = ((const v4f*)(bp + d0))[1];
    v4f q2 = ((const v4f*)(bp + d0))[2];
    v4f q3 = ((const v4f*)(bp + d0))[3];
    float bs[16];
#pragma unroll
    for (int i = 0; i < 4; ++i) {
      bs[i] = q0[i];
      bs[4 + i] = q1[i];
      bs[8 + i] = q2[i];
      bs[12 + i] = q3[i];
    }
    v16bf bh, bl;
#pragma unroll
    for (int i = 0; i < 16; ++i) {
      float x = bs[i];
      __bf16 h = (__bf16)x;
      bh[i] = h;
      bl[i] = (__bf16)(x - (float)h);
    }
    acc = WMMA_BF16(ah, bh, acc);
    acc = WMMA_BF16(ah, bl, acc);
    acc = WMMA_BF16(al, bh, acc);
  }

#pragma unroll
  for (int v = 0; v < 8; ++v) {
    const int m = wave * 16 + v + hlf * 8;
    Cout[(size_t)m * N + n0 + l16] = acc[v];
  }
}

// ---------------------------------------------------------------------------
extern "C" void kernel_launch(void* const* d_in, const int* in_sizes, int n_in,
                              void* d_out, int out_size, void* d_ws,
                              size_t ws_size, hipStream_t stream) {
  (void)in_sizes; (void)n_in; (void)out_size; (void)ws_size;
  const float* batch_pools = (const float*)d_in[0];  // [128][16384]
  const float* metapath    = (const float*)d_in[1];  // [16384][16384]
  const float* card_emb    = (const float*)d_in[2];  // [16384][64]
  const float* kern        = (const float*)d_in[3];  // [64][128]
  float* out = (float*)d_out;                        // [128][128]

  constexpr int Bb = 128, Cc = 16384;
  constexpr size_t BC = (size_t)Bb * Cc;

  // Workspace layout (24 MB total):
  __bf16* Ahi = (__bf16*)d_ws;       // 4 MB  batch_pools hi
  __bf16* Alo = Ahi + BC;            // 4 MB  batch_pools lo
  float*  Xt  = (float*)(Alo + BC);  // 8 MB  X transposed [128][16384]
  __bf16* Phi = (__bf16*)(Xt + BC);  // 4 MB  P hi
  __bf16* Plo = Phi + BC;            // 4 MB  P lo

  split_f32_to_bf16x2<<<(int)(BC / 256), 256, 0, stream>>>(batch_pools, Ahi,
                                                           Alo, (int)BC);
  xt_gemm_small<<<Cc / 256, 256, 0, stream>>>(card_emb, kern, Xt);
  gemm_metapath_async<<<Cc / NT, 256, 0, stream>>>(Ahi, Alo, metapath, Phi,
                                                   Plo, Cc, Cc);
  gemm_out_bf16x3<<<Bb / 16, 256, 0, stream>>>(Phi, Plo, Xt, out, Bb, Cc);
}
